// GATConv_34050500722693
// MI455X (gfx1250) — matrix operand order repaired and checked by reading, not compiled, not run on hardware
//
#include <hip/hip_runtime.h>
#include <hip/hip_bf16.h>
#include <math.h>

// GATConv for MI455X (gfx1250, wave32, WMMA).
// N=100000, E=1600000, IN_DIM=128, H=4, D=32 (H*D=128).

typedef __attribute__((ext_vector_type(16))) _Float16 v16h;
typedef __attribute__((ext_vector_type(8)))  float    v8f;

// ---- WMMA 16x16x32 f16 fragment index helpers (wave32) ----
// A (16x32, MxK): lane m = lane&15. VGPR v(0..3): K=2v,2v+1 (+8 if lane>=16);
//                 VGPR v(4..7): K=16+2(v-4),... (+8 if lane>=16).
// Per lane this is two contiguous 8-halfword runs -> 2x ds_load_b128.
__device__ __forceinline__ int a_frag_k(int el, int lane) {
  int v = el >> 1;
  int base = (v < 4) ? (2 * v) : (16 + 2 * (v - 4));
  return base + ((lane & 16) ? 8 : 0) + (el & 1);
}
// B (32x16, KxN): lane n = lane&15; same K order as A: el 0-7 -> K=el,
// el 8-15 -> K=16+(el-8), +8 if lane>=16.
__device__ __forceinline__ int b_frag_k(int el, int lane) {
  return (el & 7) + ((el & 8) ? 16 : 0) + ((lane & 16) ? 8 : 0);
}

// Monotonic float <-> sortable-uint key (for atomicMax-based segment max).
__device__ __forceinline__ unsigned f2key(float f) {
  unsigned u = __float_as_uint(f);
  return u ^ ((u & 0x80000000u) ? 0xFFFFFFFFu : 0x80000000u);
}
__device__ __forceinline__ float key2f(unsigned u) {
  u ^= ((u & 0x80000000u) ? 0x80000000u : 0xFFFFFFFFu);
  return __uint_as_float(u);
}

// ---------------------------------------------------------------------------
// Pack a 128x128 f32 weight matrix into per-lane B-fragment layout (f16):
//   Wp[((cb*4 + kk)*32 + lane)*16 + el] = W[(kk*32 + b_frag_k(el,lane))*128
//                                           + cb*16 + (lane&15)]
// 32 contiguous bytes per (cb,kk,lane) -> B fragment = 2x global_load_b128.
// 1024 threads total.
// ---------------------------------------------------------------------------
__global__ __launch_bounds__(256) void gat_pack_w(
    const float* __restrict__ W, _Float16* __restrict__ Wp) {
  int t = blockIdx.x * blockDim.x + threadIdx.x;  // 0..1023
  int cb   = t >> 7;
  int kk   = (t >> 5) & 3;
  int lane = t & 31;
  _Float16* d = Wp + (size_t)t * 16;
  #pragma unroll
  for (int el = 0; el < 16; ++el) {
    int k = kk * 32 + b_frag_k(el, lane);
    d[el] = (_Float16)W[(size_t)k * 128 + cb * 16 + (lane & 15)];
  }
}

// ---------------------------------------------------------------------------
// Kernel 1: fused dense projections via WMMA.
//   xl  = x @ W_l                      [N,128] f32   (workspace)
//   out = x @ W_r + bias               [N,128] f32   (d_out, init for scatter)
//   a1  = x @ a1w, a2 = x @ a2w        [N,4]   f32   (workspace)
// Block = 256 threads = 8 waves; block tile = 64 rows x 128 cols.
// Wave w owns col block w*16 and iterates 4 row tiles, reusing its B
// fragments (held in registers) across all 4 -> 32 WMMAs/wave, 4x less
// weight traffic, and independent accumulators (no WMMA RAW hazards).
// ---------------------------------------------------------------------------
__global__ __launch_bounds__(256) void gat_gemm_wmma(
    const float* __restrict__ x,
    const _Float16* __restrict__ Wlp, const _Float16* __restrict__ Wrp,
    const float* __restrict__ a1w, const float* __restrict__ a2w,
    const float* __restrict__ bias,
    float* __restrict__ xl, float* __restrict__ out,
    float* __restrict__ a1, float* __restrict__ a2, int n) {
  __shared__ _Float16 xs[64 * 128];  // 16 KB f16 tile of x
  const int tid  = threadIdx.x;
  const int row0 = blockIdx.x * 64;

  // Stage x tile: 64x128 f32 -> f16, float4 granularity (8 iters/thread).
  #pragma unroll
  for (int i = 0; i < 8; ++i) {
    int idx = tid + i * 256;   // float4 index, 2048 total
    int r  = idx >> 5;         // 32 float4 per row
    int k4 = idx & 31;
    float4 v = make_float4(0.f, 0.f, 0.f, 0.f);
    if (row0 + r < n) v = ((const float4*)x)[(size_t)(row0 + r) * 32 + k4];
    _Float16* d = &xs[r * 128 + k4 * 4];
    d[0] = (_Float16)v.x; d[1] = (_Float16)v.y;
    d[2] = (_Float16)v.z; d[3] = (_Float16)v.w;
  }
  __syncthreads();

  const int lane = tid & 31;
  const int wave = tid >> 5;
  const int c0   = wave * 16;
  const int am   = lane & 15;
  const int bn   = lane & 15;

  v8f cl[4] = {{0}, {0}, {0}, {0}};
  v8f cr[4] = {{0}, {0}, {0}, {0}};
  #pragma unroll
  for (int kk = 0; kk < 4; ++kk) {
    // Pre-packed B fragments: 32 contiguous bytes per lane.
    v16h bl = *(const v16h*)(Wlp + (((size_t)wave * 4 + kk) * 32 + lane) * 16);
    v16h br = *(const v16h*)(Wrp + (((size_t)wave * 4 + kk) * 32 + lane) * 16);
    #pragma unroll
    for (int mt = 0; mt < 4; ++mt) {
      v16h a;
      #pragma unroll
      for (int el = 0; el < 16; ++el)
        a[el] = xs[(mt * 16 + am) * 128 + kk * 32 + a_frag_k(el, lane)];
      cl[mt] = __builtin_amdgcn_wmma_f32_16x16x32_f16(
          false, a, false, bl, (short)0, cl[mt], false, false);
      cr[mt] = __builtin_amdgcn_wmma_f32_16x16x32_f16(
          false, a, false, br, (short)0, cr[mt], false, false);
    }
  }

  // C/D layout: VGPR r, lane l -> row r + 8*(l>=16), col l&15.
  const int   mhi = (lane & 16) ? 8 : 0;
  const float bv  = bias[c0 + bn];
  #pragma unroll
  for (int mt = 0; mt < 4; ++mt) {
    #pragma unroll
    for (int r = 0; r < 8; ++r) {
      int rr = row0 + mt * 16 + r + mhi;
      if (rr < n) {
        xl [(size_t)rr * 128 + c0 + bn] = cl[mt][r];
        out[(size_t)rr * 128 + c0 + bn] = cr[mt][r] + bv;  // x_r + bias
      }
    }
  }

  // a1/a2: 64 rows x 4 heads x 2 projections = 512 dots of length 128.
  #pragma unroll
  for (int i = 0; i < 2; ++i) {
    int t = tid + i * 256;   // 0..511
    int r = t >> 3;
    int which = (t >> 2) & 1;
    int h = t & 3;
    const float* aw = which ? a2w : a1w;
    float acc = 0.0f;
    #pragma unroll 8
    for (int k = 0; k < 128; ++k)
      acc += (float)xs[r * 128 + k] * aw[(size_t)k * 4 + h];
    int rr = row0 + r;
    if (rr < n) {
      if (which) a2[(size_t)rr * 4 + h] = acc;
      else       a1[(size_t)rr * 4 + h] = acc;
    }
  }
}

// Kernel 0: reset per-node softmax state each call (deterministic replay).
__global__ void gat_init(unsigned* __restrict__ mkey, float* __restrict__ s,
                         int count) {
  int i = blockIdx.x * blockDim.x + threadIdx.x;
  if (i < count) { mkey[i] = 0u; s[i] = 0.0f; }
}

// Kernel 2: per-edge logits (leaky-relu 0.2) + segment max via keyed atomicMax.
__global__ __launch_bounds__(256) void gat_edge_logits(
    const int* __restrict__ row, const int* __restrict__ col,
    const float* __restrict__ a1, const float* __restrict__ a2,
    float* __restrict__ ebuf, unsigned* __restrict__ mkey, int ne) {
  int e = blockIdx.x * blockDim.x + threadIdx.x;
  if (e >= ne) return;
  int r = row[e], c = col[e];
  float4 v1 = ((const float4*)a1)[r];
  float4 v2 = ((const float4*)a2)[c];
  float l0 = v1.x + v2.x; l0 = (l0 > 0.0f) ? l0 : 0.2f * l0;
  float l1 = v1.y + v2.y; l1 = (l1 > 0.0f) ? l1 : 0.2f * l1;
  float l2 = v1.z + v2.z; l2 = (l2 > 0.0f) ? l2 : 0.2f * l2;
  float l3 = v1.w + v2.w; l3 = (l3 > 0.0f) ? l3 : 0.2f * l3;
  ((float4*)ebuf)[e] = make_float4(l0, l1, l2, l3);
  unsigned* mk = mkey + (size_t)r * 4;
  atomicMax(mk + 0, f2key(l0));
  atomicMax(mk + 1, f2key(l1));
  atomicMax(mk + 2, f2key(l2));
  atomicMax(mk + 3, f2key(l3));
}

// Kernel 3: e = exp(logit - max[row]); accumulate segment sums; ebuf <- e.
__global__ __launch_bounds__(256) void gat_edge_exp(
    const int* __restrict__ row, const unsigned* __restrict__ mkey,
    float* __restrict__ ebuf, float* __restrict__ s, int ne) {
  int e = blockIdx.x * blockDim.x + threadIdx.x;
  if (e >= ne) return;
  int r = row[e];
  float4 lg = ((const float4*)ebuf)[e];
  const unsigned* mk = mkey + (size_t)r * 4;
  float e0 = __expf(lg.x - key2f(mk[0]));
  float e1 = __expf(lg.y - key2f(mk[1]));
  float e2 = __expf(lg.z - key2f(mk[2]));
  float e3 = __expf(lg.w - key2f(mk[3]));
  ((float4*)ebuf)[e] = make_float4(e0, e1, e2, e3);
  float* sr = s + (size_t)r * 4;
  atomicAdd(sr + 0, e0);
  atomicAdd(sr + 1, e1);
  atomicAdd(sr + 2, e2);
  atomicAdd(sr + 3, e3);
}

// Kernel 4: out[row, j] += (e/s[row,h]) * xl[col, j].  128 threads/edge,
// j = h*32+d fully coalesced (512B/edge); atomics + gathers resolve in the
// 192 MB L2 (out + xl + ebuf + indices ~150 MB working set).
__global__ __launch_bounds__(256) void gat_scatter(
    const int* __restrict__ row, const int* __restrict__ col,
    const float* __restrict__ ebuf, const float* __restrict__ s,
    const float* __restrict__ xl, float* __restrict__ out, int ne) {
  int t = blockIdx.x * blockDim.x + threadIdx.x;
  int e = t >> 7;
  int j = t & 127;
  if (e >= ne) return;
  int r = row[e], c = col[e];
  int h = j >> 5;
  float w = ebuf[(size_t)e * 4 + h] / s[(size_t)r * 4 + h];
  atomicAdd(&out[(size_t)r * 128 + j], w * xl[(size_t)c * 128 + j]);
}

// ---------------------------------------------------------------------------
extern "C" void kernel_launch(void* const* d_in, const int* in_sizes, int n_in,
                              void* d_out, int out_size, void* d_ws,
                              size_t ws_size, hipStream_t stream) {
  const float* x    = (const float*)d_in[0];
  const int*   row  = (const int*)  d_in[1];
  const int*   col  = (const int*)  d_in[2];
  const float* Wl   = (const float*)d_in[3];
  const float* Wr   = (const float*)d_in[4];
  const float* a1w  = (const float*)d_in[5];
  const float* a2w  = (const float*)d_in[6];
  const float* bias = (const float*)d_in[7];
  float* out = (float*)d_out;

  const int n  = in_sizes[0] / 128;  // N = 100000
  const int ne = in_sizes[1];        // E = 1600000

  // Workspace layout (floats), ~83 MB total:
  //   xl[N*128] | a1[N*4] | a2[N*4] | mkey[N*4](u32) | s[N*4] | ebuf[E*4]
  //   | Wlp[16384 f16] | Wrp[16384 f16]
  float*    xl   = (float*)d_ws;
  float*    a1   = xl + (size_t)n * 128;
  float*    a2   = a1 + (size_t)n * 4;
  unsigned* mkey = (unsigned*)(a2 + (size_t)n * 4);
  float*    s    = (float*)(mkey + (size_t)n * 4);
  float*    ebuf = s + (size_t)n * 4;
  _Float16* wlp  = (_Float16*)(ebuf + (size_t)ne * 4);
  _Float16* wrp  = wlp + 128 * 128;

  const int nh = n * 4;
  gat_init<<<(nh + 255) / 256, 256, 0, stream>>>(mkey, s, nh);

  gat_pack_w<<<4, 256, 0, stream>>>(Wl, wlp);
  gat_pack_w<<<4, 256, 0, stream>>>(Wr, wrp);

  gat_gemm_wmma<<<(n + 63) / 64, 256, 0, stream>>>(
      x, wlp, wrp, a1w, a2w, bias, xl, out, a1, a2, n);

  gat_edge_logits<<<(ne + 255) / 256, 256, 0, stream>>>(
      row, col, a1, a2, ebuf, mkey, ne);

  gat_edge_exp<<<(ne + 255) / 256, 256, 0, stream>>>(row, mkey, ebuf, s, ne);

  const int total = ne * 128;  // 204.8M threads, fits in int
  gat_scatter<<<(total + 255) / 256, 256, 0, stream>>>(
      row, col, ebuf, s, xl, out, ne);
}
